// MPNN_39470749450525
// MI455X (gfx1250) — matrix operand order repaired and checked by reading, not compiled, not run on hardware
//
#include <hip/hip_runtime.h>
#include <hip/hip_bf16.h>

#define N_NODES 50000
#define N_EDGES 800000
#define D_NODE 64
#define D_EDGE 32
#define MSG_DIM 160            // 2*D_NODE + D_EDGE
#define HID_DIM 224            // MSG_DIM + D_NODE
#define ALPHA 0.01f

typedef __attribute__((ext_vector_type(16))) __bf16 v16bf;
typedef __attribute__((ext_vector_type(8)))  float  v8f;

// ---- workspace layout ----
// [0, 32MB)            : msg_sum f32  [N_NODES, MSG_DIM]
// [32MB, +51.2KB)      : W_msg bf16   [MSG_DIM, MSG_DIM]  row-major (n,k)
// [.., +100.4KB)       : W_hid bf16   [HID_DIM, HID_DIM]  row-major (n,k)
#define MS_BYTES ((size_t)N_NODES * MSG_DIM * sizeof(float))
#define WM_OFF   MS_BYTES
#define WM_BYTES (MSG_DIM * MSG_DIM * (int)sizeof(__bf16))   // 51200
#define WH_OFF   (WM_OFF + (size_t)WM_BYTES)
#define WH_BYTES (HID_DIM * HID_DIM * (int)sizeof(__bf16))   // 100352

// -------- weight fp32 -> bf16 conversion (once per launch) --------
__global__ void wcvt_kernel(const float* __restrict__ Wm,
                            const float* __restrict__ Wh,
                            __bf16* __restrict__ wm_bf,
                            __bf16* __restrict__ wh_bf) {
    int i = blockIdx.x * blockDim.x + threadIdx.x;
    if (i < MSG_DIM * MSG_DIM) wm_bf[i] = (__bf16)Wm[i];
    if (i < HID_DIM * HID_DIM) wh_bf[i] = (__bf16)Wh[i];
}

__device__ __forceinline__ void cvt8(const float* __restrict__ p, v16bf& a, int j) {
    const float4 f0 = ((const float4*)p)[0];
    const float4 f1 = ((const float4*)p)[1];
    a[j + 0] = (__bf16)f0.x; a[j + 1] = (__bf16)f0.y;
    a[j + 2] = (__bf16)f0.z; a[j + 3] = (__bf16)f0.w;
    a[j + 4] = (__bf16)f1.x; a[j + 5] = (__bf16)f1.y;
    a[j + 6] = (__bf16)f1.z; a[j + 7] = (__bf16)f1.w;
}

// Stage `bytes` of weights into LDS (offset 0 of dynamic LDS) with CDNA5
// async global->LDS copies (16B per lane, ASYNCcnt-tracked, no VGPR bounce).
__device__ __forceinline__ void stage_weights_async(const __bf16* __restrict__ gsrc,
                                                    int bytes) {
    for (int off = threadIdx.x * 16; off < bytes; off += 256 * 16) {
        unsigned lds_off = (unsigned)off;    // LDS destination byte offset
        unsigned g_off   = (unsigned)off;    // GVS: SADDR64 + VADDR32
        asm volatile("global_load_async_to_lds_b128 %0, %1, %2"
                     :: "v"(lds_off), "v"(g_off), "s"(gsrc) : "memory");
    }
    asm volatile("s_wait_asynccnt 0" ::: "memory");
    __syncthreads();
}

// -------- edge message GEMM + scatter-sum --------
// One wave per 16-edge tile. msg = leaky_relu([h_n[src]|h_n[dst]|h_e] @ Wm^T + b)
// then atomicAdd into msg_sum[dst].  Wm lives in LDS for the whole block.
__global__ __launch_bounds__(256) void edge_msg_kernel(
    const float* __restrict__ h_n, const float* __restrict__ h_e,
    const int*   __restrict__ src, const int*  __restrict__ dst,
    const __bf16* __restrict__ Wm, const float* __restrict__ b_msg,
    float* __restrict__ msg_sum)
{
    extern __shared__ __bf16 lds_w[];                // [MSG_DIM][MSG_DIM]
    stage_weights_async(Wm, WM_BYTES);               // all waves, before any exit

    const int lane = threadIdx.x & 31;
    const int wave = threadIdx.x >> 5;
    const int tile = blockIdx.x * 8 + wave;          // exactly 50000 tiles
    const int r    = lane & 15;                      // A row / B,C,D column lane
    const int hi   = lane >> 4;                      // K-half selector
    const int e    = tile * 16 + r;

    const int se = src[e];
    const int de = dst[e];

    // prefetch next tile's slice of the only HBM-streaming operand
    __builtin_prefetch(h_e + (size_t)(e + 16) * D_EDGE, 0, 0);

    // ---- A fragments: 16x160 bf16, 5 chunks of K=32 ----
    // A layout: lanes 0-15 K={kb..kb+15}; lanes 16-31 K={kb+16..kb+31}.
    v16bf a[5];
    const int base0 = hi * 16;
    #pragma unroll
    for (int kc = 0; kc < 5; ++kc) {
        const int kb = kc * 32;
        #pragma unroll
        for (int c = 0; c < 2; ++c) {
            const int k = kb + base0 + c * 8;        // 8-float chunk (region-uniform)
            const float* p;
            if (k < 64)       p = h_n + (size_t)se * D_NODE + k;
            else if (k < 128) p = h_n + (size_t)de * D_NODE + (k - 64);
            else              p = h_e + (size_t)e  * D_EDGE + (k - 128);
            cvt8(p, a[kc], c * 8);
        }
    }

    // ---- WMMA: 10 output tiles of N=16, f32 accumulate, B from LDS ----
    v8f acc[10] = {};
    const int koff = hi * 16;                        // B layout: lane hi -> K upper half
    #pragma unroll
    for (int nt = 0; nt < 10; ++nt) {
        const __bf16* wrow = lds_w + (nt * 16 + r) * MSG_DIM + koff;
        #pragma unroll
        for (int kc = 0; kc < 5; ++kc) {
            const v16bf b = *(const v16bf*)(wrow + kc * 32);
            acc[nt] = __builtin_amdgcn_wmma_f32_16x16x32_bf16(
                false, a[kc], false, b, (short)0, acc[nt], false, false);
        }
    }

    // destination rows for this lane's C elements (C row m = i + 8*hi)
    int drow[8];
    #pragma unroll
    for (int i = 0; i < 8; ++i) drow[i] = dst[tile * 16 + hi * 8 + i];

    // ---- epilogue: bias + leaky relu + scatter atomic add (L2-resident) ----
    #pragma unroll
    for (int nt = 0; nt < 10; ++nt) {
        const int n = nt * 16 + r;
        const float bn = b_msg[n];
        #pragma unroll
        for (int i = 0; i < 8; ++i) {
            float v = acc[nt][i] + bn;
            v = v > 0.f ? v : ALPHA * v;
            atomicAdd(msg_sum + (size_t)drow[i] * MSG_DIM + n, v);
        }
    }
}

// -------- node update GEMM --------
// h_out = leaky_relu([msg_sum | h_n] @ Wh^T + b), Wh staged in LDS.
__global__ __launch_bounds__(256) void node_update_kernel(
    const float* __restrict__ h_n, const float* __restrict__ msg_sum,
    const __bf16* __restrict__ Wh, const float* __restrict__ b_hid,
    float* __restrict__ h_out)
{
    extern __shared__ __bf16 lds_w[];                // [HID_DIM][HID_DIM]
    stage_weights_async(Wh, WH_BYTES);               // all waves participate

    const int lane = threadIdx.x & 31;
    const int wave = threadIdx.x >> 5;
    const int tile = blockIdx.x * 8 + wave;
    if (tile >= N_NODES / 16) return;                // wave-uniform, after barrier
    const int r    = lane & 15;
    const int hi   = lane >> 4;
    const int row  = tile * 16 + r;

    // ---- A fragments: 16x224 bf16, 7 chunks of K=32 ----
    v16bf a[7];
    const int base0 = hi * 16;
    #pragma unroll
    for (int kc = 0; kc < 7; ++kc) {
        const int kb = kc * 32;
        #pragma unroll
        for (int c = 0; c < 2; ++c) {
            const int k = kb + base0 + c * 8;
            const float* p = (k < MSG_DIM)
                ? (msg_sum + (size_t)row * MSG_DIM + k)
                : (h_n     + (size_t)row * D_NODE + (k - MSG_DIM));
            cvt8(p, a[kc], c * 8);
        }
    }

    // ---- WMMA: 14 output tiles of N=16, B from LDS ----
    v8f acc[14] = {};
    const int koff = hi * 16;
    #pragma unroll
    for (int nt = 0; nt < 14; ++nt) {
        const __bf16* wrow = lds_w + (nt * 16 + r) * HID_DIM + koff;
        #pragma unroll
        for (int kc = 0; kc < 7; ++kc) {
            const v16bf b = *(const v16bf*)(wrow + kc * 32);
            acc[nt] = __builtin_amdgcn_wmma_f32_16x16x32_bf16(
                false, a[kc], false, b, (short)0, acc[nt], false, false);
        }
    }

    // ---- epilogue: bias + leaky relu + store ----
    #pragma unroll
    for (int nt = 0; nt < 14; ++nt) {
        const int n = nt * 16 + r;
        const float bn = b_hid[n];
        #pragma unroll
        for (int i = 0; i < 8; ++i) {
            const int node = tile * 16 + hi * 8 + i;   // C/D row for element i
            float v = acc[nt][i] + bn;
            v = v > 0.f ? v : ALPHA * v;
            h_out[(size_t)node * HID_DIM + n] = v;
        }
    }
}

extern "C" void kernel_launch(void* const* d_in, const int* in_sizes, int n_in,
                              void* d_out, int out_size, void* d_ws, size_t ws_size,
                              hipStream_t stream) {
    const float* h_n   = (const float*)d_in[0];
    const float* h_e   = (const float*)d_in[1];
    const int*   src   = (const int*)  d_in[2];
    const int*   dst   = (const int*)  d_in[3];
    const float* W_msg = (const float*)d_in[4];
    const float* b_msg = (const float*)d_in[5];
    const float* W_hid = (const float*)d_in[6];
    const float* b_hid = (const float*)d_in[7];
    float* out = (float*)d_out;

    char*   ws      = (char*)d_ws;
    float*  msg_sum = (float*)ws;
    __bf16* wm_bf   = (__bf16*)(ws + WM_OFF);
    __bf16* wh_bf   = (__bf16*)(ws + WH_OFF);

    // zero the scatter accumulator each launch (deterministic, graph-capturable)
    hipMemsetAsync(msg_sum, 0, MS_BYTES, stream);

    wcvt_kernel<<<(HID_DIM * HID_DIM + 255) / 256, 256, 0, stream>>>(
        W_msg, W_hid, wm_bf, wh_bf);

    // 800000 edges / 16 rows per tile / 8 waves per block = 6250 blocks (exact)
    edge_msg_kernel<<<N_EDGES / 16 / 8, 256, WM_BYTES, stream>>>(
        h_n, h_e, src, dst, wm_bf, b_msg, msg_sum);

    // 3125 node tiles / 8 waves per block
    node_update_kernel<<<(N_NODES / 16 + 7) / 8, 256, WH_BYTES, stream>>>(
        h_n, msg_sum, wh_bf, b_hid, out);
}
